// HGTLayer_22179211116713
// MI455X (gfx1250) — matrix operand order repaired and checked by reading, not compiled, not run on hardware
//
#include <hip/hip_runtime.h>
#include <hip/hip_bf16.h>
#include <math.h>

// ---------------------------------------------------------------------------
// HGT layer for MI455X (gfx1250, wave32, WMMA, async-to-LDS)
// N=50000, E=1600000, D=128, H=8, DK=16, R=8
// ---------------------------------------------------------------------------

#define NN      50000
#define EE      1600000
#define DIM     128
#define HH      8
#define DKK     16
#define RR      8
#define NSEG    (NN * RR)           // 400000 segments
#define NSEGH   (NSEG * HH)         // per (seg, head)
#define ROWS    80                  // node rows per GEMM block (5 tiles of 16)
#define NBLK    (NN / ROWS)         // 625

typedef __bf16 v16bf __attribute__((ext_vector_type(16)));
typedef float  v8f   __attribute__((ext_vector_type(8)));

// float atomic max via signed/unsigned integer atomics (standard bit trick)
__device__ __forceinline__ void atomicMaxF(float* addr, float val) {
    if (val >= 0.0f) {
        atomicMax((int*)addr, __float_as_int(val));
    } else {
        atomicMin((unsigned int*)addr, __float_as_uint(val));
    }
}

// generic -> LDS byte offset (for async-to-LDS VDST operand)
__device__ __forceinline__ unsigned lds_offset(const void* p) {
    return (unsigned)(unsigned long long)(const __attribute__((address_space(3))) void*)p;
}

__device__ __forceinline__ void async_load_b128(unsigned lds_off, const void* gaddr) {
    asm volatile("global_load_async_to_lds_b128 %0, %1, off"
                 :: "v"(lds_off), "v"(gaddr) : "memory");
}

__device__ __forceinline__ void wait_asynccnt0() {
#if __has_builtin(__builtin_amdgcn_s_wait_asynccnt)
    __builtin_amdgcn_s_wait_asynccnt(0);
#else
    asm volatile("s_wait_asynccnt 0x0" ::: "memory");
#endif
}

// ---------------------------------------------------------------------------
// K0a: init + weight fragment pre-swizzle
// Weight fragment layout (WMMA B operand, wave32):
//   Wfrag[((strip*4 + chunk)*32 + lane)*16 + i]
//     = W[o][kk],  o  = strip*16 + (lane&15)
//                  kk = chunk*32 + ((lane<16)?0:8) + ((i<8)?i:i+8)
// Each lane's 16 bf16 (32 bytes) are contiguous -> 2x global_load_b128.
// ---------------------------------------------------------------------------
__global__ void init_ws_kernel(const float* __restrict__ Wq, const float* __restrict__ Wk,
                               const float* __restrict__ Wv, const float* __restrict__ Wa,
                               __bf16* __restrict__ WqF, __bf16* __restrict__ WkF,
                               __bf16* __restrict__ WvF, __bf16* __restrict__ WaF,
                               float* __restrict__ t, float* __restrict__ seg_max,
                               float* __restrict__ seg_den, int* __restrict__ seg_cnt) {
    const int tid    = blockIdx.x * blockDim.x + threadIdx.x;
    const int stride = gridDim.x * blockDim.x;
    for (int i = tid; i < NN * DIM; i += stride) t[i] = 0.0f;
    for (int i = tid; i < NSEGH; i += stride) { seg_max[i] = -INFINITY; seg_den[i] = 0.0f; }
    for (int i = tid; i < NSEG; i += stride) seg_cnt[i] = 0;
    for (int f = tid; f < DIM * DIM; f += stride) {
        const int i     = f & 15;
        const int lane  = (f >> 4) & 31;
        const int chunk = (f >> 9) & 3;
        const int strip = f >> 11;
        const int o  = strip * 16 + (lane & 15);
        const int kk = chunk * 32 + ((lane < 16) ? 0 : 8) + ((i < 8) ? i : i + 8);
        const int src = o * DIM + kk;
        WqF[f] = (__bf16)Wq[src];
        WkF[f] = (__bf16)Wk[src];
        WvF[f] = (__bf16)Wv[src];
        WaF[f] = (__bf16)Wa[src];
    }
}

// ---------------------------------------------------------------------------
// K0b: convert activations f32 -> bf16 (so GEMM tiles can be async-DMA'd raw)
// ---------------------------------------------------------------------------
__global__ void convert_bf16_kernel(const float* __restrict__ src_h,
                                    const float* __restrict__ dst_h,
                                    __bf16* __restrict__ src_bf,
                                    __bf16* __restrict__ dst_bf) {
    const int tid    = blockIdx.x * blockDim.x + threadIdx.x;
    const int stride = gridDim.x * blockDim.x;
    for (int i = tid; i < NN * DIM; i += stride) {
        src_bf[i] = (__bf16)src_h[i];
        dst_bf[i] = (__bf16)dst_h[i];
    }
}

// ---------------------------------------------------------------------------
// WMMA A-fragment from LDS tile (16 rows x 128 cols bf16), ISA 7.12.2 layout:
//   lane L -> row L%16; elems 0..7 = K base..base+7, elems 8..15 = K base+16..+23,
//   base = chunk*32 + ((L<16)?0:8)   -> two ds_load_b128 per fragment
// ---------------------------------------------------------------------------
__device__ __forceinline__ v16bf load_fragA_lds(const __bf16* xs, int lane, int kchunk) {
    const int row = lane & 15;
    const int ka  = kchunk * 32 + ((lane < 16) ? 0 : 8);
    const __bf16* rp = xs + row * DIM + ka;
    v16bf a;
#pragma unroll
    for (int i = 0; i < 8; ++i) {
        a[i]     = rp[i];
        a[8 + i] = rp[16 + i];
    }
    return a;
}

// ---------------------------------------------------------------------------
// K1: dense projection  Y[N,128] = X[N,128] @ Wᵀ + bias   (bf16 WMMA, f32 acc)
// Block = 80-node tile (5 x 16-row WMMA tiles), 8 waves = 8 column strips.
// x-tile staged into LDS with GLOBAL_LOAD_ASYNC_TO_LDS_B128 (ASYNCcnt).
// B-fragments held in registers, reused across the 5 row tiles (20 WMMA/wave).
// ---------------------------------------------------------------------------
__global__ __launch_bounds__(256) void proj_gemm_kernel(const __bf16* __restrict__ Xbf,
                                                        const __bf16* __restrict__ Wfrag,
                                                        const float* __restrict__ bias,
                                                        float* __restrict__ Y) {
    __shared__ __bf16 xs[ROWS * DIM];           // 20 KB
    const int base = blockIdx.x * ROWS;
    const __bf16* gp0 = Xbf + (size_t)base * DIM;
#pragma unroll
    for (int rnd = 0; rnd < (ROWS * DIM) / (256 * 8); ++rnd) {   // 5 rounds x 16B/thread
        const int idx = rnd * 2048 + threadIdx.x * 8;
        async_load_b128(lds_offset(xs + idx), gp0 + idx);
    }
    wait_asynccnt0();
    __syncthreads();

    const int lane    = threadIdx.x & 31;
    const int wv      = threadIdx.x >> 5;
    const int colBase = wv * 16;
    const int col     = colBase + (lane & 15);

    // one 32-byte contiguous fragment load per chunk
    v16bf b[4];
#pragma unroll
    for (int ch = 0; ch < 4; ++ch)
        b[ch] = *(const v16bf*)(Wfrag + (size_t)((wv * 4 + ch) * 32 + lane) * 16);

    const float bc     = bias[col];
    const int   rowOff = (lane < 16) ? 0 : 8;

#pragma unroll
    for (int tile = 0; tile < ROWS / 16; ++tile) {
        v8f acc = {};
#pragma unroll
        for (int ch = 0; ch < 4; ++ch) {
            v16bf a = load_fragA_lds(xs + tile * 16 * DIM, lane, ch);
            acc = __builtin_amdgcn_wmma_f32_16x16x32_bf16(false, a, false, b[ch],
                                                          (short)0, acc, false, false);
        }
#pragma unroll
        for (int j = 0; j < 8; ++j)
            Y[(size_t)(base + tile * 16 + j + rowOff) * DIM + col] = acc[j] + bc;
    }
}

// ---------------------------------------------------------------------------
// K2: per-edge attention logits. One wave per edge.
// lane -> (h = lane/4, quarter j = lane%4 owning 4 columns of DK)
// att[e,h] = (q[dst,h,:] . (k[src,h,:] @ rel_att[r,h])) * pri[r,h] / sqrt(DK)
// ---------------------------------------------------------------------------
__global__ __launch_bounds__(256) void edge_att_kernel(const float* __restrict__ q,
                                                       const float* __restrict__ k,
                                                       const float* __restrict__ rel_att,
                                                       const float* __restrict__ rel_pri,
                                                       const int* __restrict__ src_idx,
                                                       const int* __restrict__ dst_idx,
                                                       const int* __restrict__ edge_type,
                                                       float* __restrict__ att,
                                                       float* __restrict__ seg_max,
                                                       int* __restrict__ seg_cnt) {
    const int e = (int)((blockIdx.x * blockDim.x + threadIdx.x) >> 5);
    if (e >= EE) return;
    const int lane = threadIdx.x & 31;
    const int h = lane >> 2, jq = lane & 3;

    const int s = src_idx[e], d = dst_idx[e], r = edge_type[e];
    const float* kp = k + (size_t)s * DIM + h * DKK;
    const float* qp = q + (size_t)d * DIM + h * DKK + jq * 4;
    const float* Ap = rel_att + (size_t)(r * HH + h) * (DKK * DKK) + jq * 4;

    float key[4] = {0.f, 0.f, 0.f, 0.f};
#pragma unroll
    for (int dd = 0; dd < DKK; ++dd) {
        const float kv = kp[dd];
#pragma unroll
        for (int c = 0; c < 4; ++c) key[c] = fmaf(kv, Ap[dd * DKK + c], key[c]);
    }
    float part = 0.f;
#pragma unroll
    for (int c = 0; c < 4; ++c) part = fmaf(qp[c], key[c], part);
    part += __shfl_xor(part, 1, 32);
    part += __shfl_xor(part, 2, 32);

    if (jq == 0) {
        const float a = part * rel_pri[r * HH + h] * 0.25f;   // 1/sqrt(16)
        att[(size_t)e * HH + h] = a;
        const int seg = d * RR + r;
        atomicMaxF(&seg_max[(size_t)seg * HH + h], a);
        if (h == 0) atomicAdd(&seg_cnt[seg], 1);
    }
}

// ---------------------------------------------------------------------------
// K3: exp(att - max) + segment denominators. One thread per (edge, head).
// ---------------------------------------------------------------------------
__global__ void edge_norm_kernel(const int* __restrict__ dst_idx,
                                 const int* __restrict__ edge_type,
                                 const float* __restrict__ seg_max,
                                 float* __restrict__ att,       // in: logits, out: exp
                                 float* __restrict__ seg_den) {
    const int i = blockIdx.x * blockDim.x + threadIdx.x;
    if (i >= EE * HH) return;
    const int e = i >> 3, h = i & 7;
    const int seg = dst_idx[e] * RR + edge_type[e];
    const float ex = __expf(att[i] - seg_max[(size_t)seg * HH + h]);
    att[i] = ex;
    atomicAdd(&seg_den[(size_t)seg * HH + h], ex);
}

// ---------------------------------------------------------------------------
// K4: normalized weighted aggregation into t[dst]. One wave per edge.
// msg = v[src,h,:] @ rel_msg[r,h];  t[dst] += a * msg   (f32 global atomics)
// ---------------------------------------------------------------------------
__global__ __launch_bounds__(256) void edge_agg_kernel(const float* __restrict__ v,
                                                       const float* __restrict__ rel_msg,
                                                       const int* __restrict__ src_idx,
                                                       const int* __restrict__ dst_idx,
                                                       const int* __restrict__ edge_type,
                                                       const float* __restrict__ ex,
                                                       const float* __restrict__ seg_den,
                                                       float* __restrict__ t) {
    const int e = (int)((blockIdx.x * blockDim.x + threadIdx.x) >> 5);
    if (e >= EE) return;
    const int lane = threadIdx.x & 31;
    const int h = lane >> 2, jq = lane & 3;

    const int s = src_idx[e], d = dst_idx[e], r = edge_type[e];
    const int seg = d * RR + r;
    const float a = ex[(size_t)e * HH + h] / seg_den[(size_t)seg * HH + h];

    const float* vp = v + (size_t)s * DIM + h * DKK;
    const float* Mp = rel_msg + (size_t)(r * HH + h) * (DKK * DKK) + jq * 4;

    float msg[4] = {0.f, 0.f, 0.f, 0.f};
#pragma unroll
    for (int dd = 0; dd < DKK; ++dd) {
        const float vv = vp[dd];
#pragma unroll
        for (int c = 0; c < 4; ++c) msg[c] = fmaf(vv, Mp[dd * DKK + c], msg[c]);
    }
    float* tp = t + (size_t)d * DIM + h * DKK + jq * 4;
#pragma unroll
    for (int c = 0; c < 4; ++c) atomicAdd(&tp[c], a * msg[c]);
}

// ---------------------------------------------------------------------------
// K5: mean over present relations, output GEMM (WMMA) + sigmoid-skip blend
// ---------------------------------------------------------------------------
__global__ __launch_bounds__(256) void out_gemm_kernel(const float* __restrict__ T,
                                                       const int* __restrict__ seg_cnt,
                                                       const __bf16* __restrict__ WaF,
                                                       const float* __restrict__ ba,
                                                       const float* __restrict__ dst_h,
                                                       const float* __restrict__ skip,
                                                       float* __restrict__ out) {
    __shared__ __bf16 xs[ROWS * DIM];           // 20 KB
    __shared__ float inv_nrel[ROWS];
    const int base = blockIdx.x * ROWS;
    if (threadIdx.x < ROWS) {
        const int n = base + threadIdx.x;
        int nr = 0;
#pragma unroll
        for (int r = 0; r < RR; ++r) nr += (seg_cnt[n * RR + r] > 0) ? 1 : 0;
        inv_nrel[threadIdx.x] = 1.0f / (float)((nr > 1) ? nr : 1);
    }
    __syncthreads();
    for (int i = threadIdx.x; i < ROWS * DIM; i += 256)
        xs[i] = (__bf16)(T[(size_t)base * DIM + i] * inv_nrel[i >> 7]);
    __syncthreads();

    const int lane    = threadIdx.x & 31;
    const int wv      = threadIdx.x >> 5;
    const int colBase = wv * 16;
    const int col     = colBase + (lane & 15);

    v16bf b[4];
#pragma unroll
    for (int ch = 0; ch < 4; ++ch)
        b[ch] = *(const v16bf*)(WaF + (size_t)((wv * 4 + ch) * 32 + lane) * 16);

    const float alpha  = 1.0f / (1.0f + __expf(-skip[0]));
    const float beta   = 1.0f - alpha;
    const float bc     = ba[col];
    const int   rowOff = (lane < 16) ? 0 : 8;

#pragma unroll
    for (int tile = 0; tile < ROWS / 16; ++tile) {
        v8f acc = {};
#pragma unroll
        for (int ch = 0; ch < 4; ++ch) {
            v16bf a = load_fragA_lds(xs + tile * 16 * DIM, lane, ch);
            acc = __builtin_amdgcn_wmma_f32_16x16x32_bf16(false, a, false, b[ch],
                                                          (short)0, acc, false, false);
        }
#pragma unroll
        for (int j = 0; j < 8; ++j) {
            const size_t idx = (size_t)(base + tile * 16 + j + rowOff) * DIM + col;
            out[idx] = (acc[j] + bc) * alpha + dst_h[idx] * beta;
        }
    }
}

// ---------------------------------------------------------------------------
// launcher
// ---------------------------------------------------------------------------
extern "C" void kernel_launch(void* const* d_in, const int* in_sizes, int n_in,
                              void* d_out, int out_size, void* d_ws, size_t ws_size,
                              hipStream_t stream) {
    const float* src_h   = (const float*)d_in[0];
    const float* dst_h   = (const float*)d_in[1];
    const float* Wk      = (const float*)d_in[2];
    const float* bk      = (const float*)d_in[3];
    const float* Wq      = (const float*)d_in[4];
    const float* bq      = (const float*)d_in[5];
    const float* Wv      = (const float*)d_in[6];
    const float* bv      = (const float*)d_in[7];
    const float* Wa      = (const float*)d_in[8];
    const float* ba      = (const float*)d_in[9];
    const float* rel_pri = (const float*)d_in[10];
    const float* rel_att = (const float*)d_in[11];
    const float* rel_msg = (const float*)d_in[12];
    const float* skip    = (const float*)d_in[13];
    const int*   src_idx = (const int*)d_in[14];
    const int*   dst_idx = (const int*)d_in[15];
    const int*   edge_ty = (const int*)d_in[16];
    float*       out     = (float*)d_out;

    // workspace carve-up (aligned to 256B)
    char* p = (char*)d_ws;
    auto alloc = [&](size_t bytes) -> char* {
        char* r = p;
        p += (bytes + 255) & ~(size_t)255;
        return r;
    };
    __bf16* WqF    = (__bf16*)alloc(DIM * DIM * sizeof(__bf16));
    __bf16* WkF    = (__bf16*)alloc(DIM * DIM * sizeof(__bf16));
    __bf16* WvF    = (__bf16*)alloc(DIM * DIM * sizeof(__bf16));
    __bf16* WaF    = (__bf16*)alloc(DIM * DIM * sizeof(__bf16));
    __bf16* src_bf = (__bf16*)alloc((size_t)NN * DIM * sizeof(__bf16));
    __bf16* dst_bf = (__bf16*)alloc((size_t)NN * DIM * sizeof(__bf16));
    float* qbuf    = (float*)alloc((size_t)NN * DIM * sizeof(float));
    float* kbuf    = (float*)alloc((size_t)NN * DIM * sizeof(float));
    float* vbuf    = (float*)alloc((size_t)NN * DIM * sizeof(float));
    float* attbuf  = (float*)alloc((size_t)EE * HH * sizeof(float));
    float* seg_max = (float*)alloc((size_t)NSEGH * sizeof(float));
    float* seg_den = (float*)alloc((size_t)NSEGH * sizeof(float));
    int*   seg_cnt = (int*)alloc((size_t)NSEG * sizeof(int));
    float* tbuf    = (float*)alloc((size_t)NN * DIM * sizeof(float));

    // K0: init + weight fragment pre-swizzle; activation f32->bf16
    init_ws_kernel<<<1024, 256, 0, stream>>>(Wq, Wk, Wv, Wa, WqF, WkF, WvF, WaF,
                                             tbuf, seg_max, seg_den, seg_cnt);
    convert_bf16_kernel<<<1024, 256, 0, stream>>>(src_h, dst_h, src_bf, dst_bf);

    // K1: q/k/v projections (WMMA bf16, async-to-LDS tile staging)
    proj_gemm_kernel<<<NBLK, 256, 0, stream>>>(dst_bf, WqF, bq, qbuf);
    proj_gemm_kernel<<<NBLK, 256, 0, stream>>>(src_bf, WkF, bk, kbuf);
    proj_gemm_kernel<<<NBLK, 256, 0, stream>>>(src_bf, WvF, bv, vbuf);

    // K2: per-edge attention logits + segment max/count
    edge_att_kernel<<<EE / 8, 256, 0, stream>>>(qbuf, kbuf, rel_att, rel_pri,
                                                src_idx, dst_idx, edge_ty,
                                                attbuf, seg_max, seg_cnt);

    // K3: exp + denominators
    edge_norm_kernel<<<(EE * HH) / 256, 256, 0, stream>>>(dst_idx, edge_ty, seg_max,
                                                          attbuf, seg_den);

    // K4: normalized aggregation into t
    edge_agg_kernel<<<EE / 8, 256, 0, stream>>>(vbuf, rel_msg, src_idx, dst_idx,
                                                edge_ty, attbuf, seg_den, tbuf);

    // K5: mean over relations + output GEMM + skip blend
    out_gemm_kernel<<<NBLK, 256, 0, stream>>>(tbuf, seg_cnt, WaF, ba, dst_h,
                                              skip, out);
}
